// CRNN_2791728742849
// MI455X (gfx1250) — compile-verified, matrix operand order checked by reading, not run on hardware
//
#include <hip/hip_runtime.h>

// Clockwork RNN for MI455X (gfx1250, wave32, WMMA).
//
// Decomposition: batch columns are independent through the recurrence, so the
// grid is 16 column-groups (16 batch cols = one WMMA N tile) x 4 row-WGs.
// Each WG keeps its bf16 weight slices LDS-resident (256 KB of the WGP's
// 320 KB LDS), pre-swizzled into WMMA A-fragment order by a prep kernel.
// H is exchanged between the 4 row-WGs of a group via a double-buffered,
// K-contiguous bf16 region in L2 with a per-group software barrier (plus the
// CDNA5 cluster barrier instruction, a NOP unless dispatched as a cluster).
// f32 H lives in registers for the exact clock-gated blend; matmuls use
// v_wmma_f32_16x16x32_bf16 with f32 accumulation. f32->bf16 conversions use
// __builtin_convertvector so the backend emits packed cvt ops (which
// co-execute with XDL WMMA) instead of scalar bit-twiddling.

#define T_STEPS 1024
#define BATCH   256
#define DIN     256
#define HID     512
#define DOUT    256

typedef __attribute__((ext_vector_type(16))) __bf16       bf16x16;
typedef __attribute__((ext_vector_type(8)))  __bf16       bf16x8;
typedef __attribute__((ext_vector_type(16))) float        f32x16;
typedef __attribute__((ext_vector_type(8)))  float        f32x8;
typedef __attribute__((ext_vector_type(4)))  float        f32x4;
typedef __attribute__((ext_vector_type(4)))  unsigned int u32x4;

struct U32x8 { u32x4 lo, hi; };
struct F32x16S { f32x4 a, b, c, d; };

static __device__ __forceinline__ bf16x16 make_frag(u32x4 lo, u32x4 hi) {
  U32x8 t; t.lo = lo; t.hi = hi;
  return __builtin_bit_cast(bf16x16, t);
}

// 16 floats (two contiguous runs of 8) -> bf16x16 fragment, RNE via packed cvt
static __device__ __forceinline__ bf16x16 cvt_frag(f32x4 a, f32x4 b, f32x4 c, f32x4 d) {
  F32x16S s; s.a = a; s.b = b; s.c = c; s.d = d;
  f32x16 v = __builtin_bit_cast(f32x16, s);
  return __builtin_convertvector(v, bf16x16);
}

static __device__ __forceinline__ unsigned short f2bf(float f) {
  unsigned u = __builtin_bit_cast(unsigned, f);
  return (unsigned short)((u + 0x7fffu + ((u >> 16) & 1u)) >> 16);  // RNE
}

static __device__ __forceinline__ f32x8 wmma_bf16(bf16x16 a, bf16x16 b, f32x8 c) {
  return __builtin_amdgcn_wmma_f32_16x16x32_bf16(false, a, false, b, (short)0, c,
                                                 false, false);
}

// ---------------------------------------------------------------------------
// Prep: swizzle weights (f32 -> bf16) into WMMA A-fragment order.
// Fragment tile = 16 rows x 32 K, 512 elems; lane L owns 16 contiguous bf16 at
// tile*512 + L*16, holding rows m = 16*rowtile + (L&15), K runs
// {k0+8*(L>>4)+0..7, k0+16+8*(L>>4)+0..7}.
// ---------------------------------------------------------------------------
static __device__ __forceinline__ unsigned short frag_elem(
    const float* __restrict__ src, unsigned e, unsigned ksteps, unsigned stride) {
  const unsigned tile    = e >> 9;
  const unsigned w       = e & 511u;
  const unsigned lane    = w >> 4;
  const unsigned idx     = w & 15u;
  const unsigned rowtile = tile / ksteps;
  const unsigned ks      = tile - rowtile * ksteps;
  const unsigned m = rowtile * 16u + (lane & 15u);
  const unsigned k = ks * 32u + ((idx >> 3) << 4) + ((lane >> 4) << 3) + (idx & 7u);
  return f2bf(src[(size_t)m * stride + k]);
}

__global__ void crnn_prep(const float* __restrict__ Wi, const float* __restrict__ Wh,
                          const float* __restrict__ Wo,
                          unsigned short* __restrict__ WhF,
                          unsigned short* __restrict__ WiF,
                          unsigned short* __restrict__ WoF,
                          float* __restrict__ biasH, float* __restrict__ biasO) {
  const unsigned n = blockIdx.x * blockDim.x + threadIdx.x;
  const unsigned NWH = 512u * 512u;  // 262144
  const unsigned NWI = 512u * 256u;  // 131072
  const unsigned NWO = 256u * 512u;  // 131072
  if (n < NWH) {
    WhF[n] = frag_elem(Wh, n, 16u, 513u);
  } else if (n < NWH + NWI) {
    WiF[n - NWH] = frag_elem(Wi, n - NWH, 8u, 257u);
  } else if (n < NWH + NWI + NWO) {
    WoF[n - NWH - NWI] = frag_elem(Wo, n - NWH - NWI, 16u, 513u);
  } else if (n < NWH + NWI + NWO + 512u) {
    const unsigned r = n - (NWH + NWI + NWO);
    biasH[r] = Wi[(size_t)r * 257u + 256u] + Wh[(size_t)r * 513u + 512u];
  } else if (n < NWH + NWI + NWO + 512u + 256u) {
    const unsigned r = n - (NWH + NWI + NWO + 512u);
    biasO[r] = Wo[(size_t)r * 513u + 512u];
  }
}

// ---------------------------------------------------------------------------
// Persistent recurrent kernel. Grid = 64 WGs: blockIdx = cg*4 + rg.
//   cg in [0,16): batch columns cg*16..+15 (one WMMA N tile)
//   rg in [0,4):  pre rows rg*128..+127, Wo rows rg*64..+63
// 256 threads = 8 waves; wave w owns pre row-tile rg*8+w (24 WMMA/step),
// waves 0..3 additionally own one Wo row-tile (16 WMMA/step).
// ---------------------------------------------------------------------------
__global__ __launch_bounds__(256, 1)
void crnn_recurrent(const float* __restrict__ X, float* __restrict__ out,
                    const unsigned short* __restrict__ WhF,
                    const unsigned short* __restrict__ WiF,
                    const unsigned short* __restrict__ WoF,
                    const float* __restrict__ biasH,
                    const float* __restrict__ biasO,
                    unsigned short* __restrict__ HT,  // 2 x [BATCH][HID] bf16
                    unsigned* __restrict__ bar) {
  extern __shared__ unsigned short smem[];   // 256 KB dynamic LDS
  unsigned short* WhS = smem;                // 128*512 bf16 = 128 KB
  unsigned short* WiS = WhS + 128 * 512;     // 128*256 bf16 =  64 KB
  unsigned short* WoS = WiS + 128 * 256;     //  64*512 bf16 =  64 KB

  const int cg   = blockIdx.x >> 2;
  const int rg   = blockIdx.x & 3;
  const int tid  = threadIdx.x;
  const int wave = tid >> 5;
  const int lane = tid & 31;
  const int lmod = lane & 15;
  const int off8 = (lane >> 4) << 3;
  const int wrow = wave & 3;

  // -- stage fragment-ordered weight slices into LDS (contiguous spans) --
  {
    const u32x4* s = (const u32x4*)(WhF + (size_t)(rg * 8 * 16) * 512);
    u32x4*       d = (u32x4*)WhS;
    for (int i = tid; i < (128 * 512 * 2) / 16; i += 256) d[i] = s[i];
    s = (const u32x4*)(WiF + (size_t)(rg * 8 * 8) * 512);
    d = (u32x4*)WiS;
    for (int i = tid; i < (128 * 256 * 2) / 16; i += 256) d[i] = s[i];
    s = (const u32x4*)(WoF + (size_t)(rg * 4 * 16) * 512);
    d = (u32x4*)WoS;
    for (int i = tid; i < (64 * 512 * 2) / 16; i += 256) d[i] = s[i];
  }
  __syncthreads();

  const int bcol    = cg * 16 + lmod;                  // this lane's batch column
  const int rowbase = (rg * 8 + wave) * 16 + off8;     // first of 8 pre rows
  const unsigned pmask = (1u << (rowbase >> 6)) - 1u;  // clock period mask
  const int obase = rg * 64 + wrow * 16 + off8;        // first of 8 Wo rows

  float bh[8], bo[8];
#pragma unroll
  for (int r = 0; r < 8; ++r) bh[r] = biasH[rowbase + r];
#pragma unroll
  for (int r = 0; r < 8; ++r) bo[r] = biasO[obase + r];

  const f32x8 zacc = {0.f, 0.f, 0.f, 0.f, 0.f, 0.f, 0.f, 0.f};
  f32x8 hprev = zacc;  // f32 H for exact clock gating (8 rows x this column)

  unsigned short* ht0 = HT + (size_t)bcol * HID;                // buffer 0
  unsigned short* ht1 = HT + (size_t)(BATCH + bcol) * HID;      // buffer 1
  unsigned* mybar = bar + cg * 16;                              // 64B-spaced ctrs

  const unsigned short* WhA = WhS + (size_t)wave * 16 * 512 + lane * 16;
  const unsigned short* WiA = WiS + (size_t)wave * 8 * 512 + lane * 16;
  const unsigned short* WoA = WoS + (size_t)wrow * 16 * 512 + lane * 16;

  for (int t = 0; t < T_STEPS; ++t) {
    const unsigned short* hcur = (t & 1) ? ht1 : ht0;
    unsigned short*       hnxt = (t & 1) ? ht0 : ht1;
    const float* xrow = X + ((size_t)t * BATCH + bcol) * DIN;

    // ---- phase 1: pre = Wi@x + Wh@H + b on own 16-row tile ----
    f32x8 acc = zacc;
    const unsigned short* a = WiA;
#pragma unroll
    for (int ks = 0; ks < 8; ++ks) {                       // K over DIN
      bf16x16 af = make_frag(*(const u32x4*)a, *(const u32x4*)(a + 8));
      a += 512;
      const int k0 = ks * 32 + off8;
      bf16x16 bf = cvt_frag(*(const f32x4*)(xrow + k0),
                            *(const f32x4*)(xrow + k0 + 4),
                            *(const f32x4*)(xrow + k0 + 16),
                            *(const f32x4*)(xrow + k0 + 20));
      acc = wmma_bf16(af, bf, acc);
    }
    a = WhA;
#pragma unroll
    for (int ks = 0; ks < 16; ++ks) {                      // K over HID
      bf16x16 af = make_frag(*(const u32x4*)a, *(const u32x4*)(a + 8));
      a += 512;
      const int k0 = ks * 32 + off8;
      u32x4 blo = *(const u32x4*)(hcur + k0);
      u32x4 bhi = *(const u32x4*)(hcur + k0 + 16);
      acc = wmma_bf16(af, make_frag(blo, bhi), acc);
    }

    // clock-gated update (mask is exactly 0/1 per 64-row group)
    if ((((unsigned)t) & pmask) == 0u) {
#pragma unroll
      for (int r = 0; r < 8; ++r) hprev[r] = tanhf(acc[r] + bh[r]);
    }
    // publish bf16 H' (K-contiguous, 16B per lane) via packed cvt
    *(bf16x8*)(hnxt + rowbase) = __builtin_convertvector(hprev, bf16x8);

    if (t + 1 < T_STEPS)
      __builtin_prefetch(X + ((size_t)(t + 1) * BATCH + bcol) * DIN, 0, 1);

    // ---- per-column-group barrier (4 row-WGs) ----
    __threadfence();
    __syncthreads();
    if (tid == 0) {
      atomicAdd(mybar, 1u);
      const unsigned tgt = (unsigned)(t + 1) * 4u;
      while (__hip_atomic_load(mybar, __ATOMIC_ACQUIRE, __HIP_MEMORY_SCOPE_AGENT) < tgt)
        __builtin_amdgcn_s_sleep(1);
    }
    __builtin_amdgcn_s_cluster_barrier();  // NOP outside a real cluster dispatch
    __syncthreads();
    __threadfence();

    // ---- phase 3: y = tanh(Wo @ H' + bo) on own Wo row tile ----
    if (wave < 4) {
      f32x8 acc2 = zacc;
      const unsigned short* a2 = WoA;
#pragma unroll
      for (int ks = 0; ks < 16; ++ks) {
        bf16x16 af = make_frag(*(const u32x4*)a2, *(const u32x4*)(a2 + 8));
        a2 += 512;
        const int k0 = ks * 32 + off8;
        u32x4 blo = *(const u32x4*)(hnxt + k0);
        u32x4 bhi = *(const u32x4*)(hnxt + k0 + 16);
        acc2 = wmma_bf16(af, make_frag(blo, bhi), acc2);
      }
      float* yp = out + ((size_t)t * BATCH + bcol) * DOUT + obase;
      f32x4 y0, y1;
#pragma unroll
      for (int r = 0; r < 4; ++r) y0[r] = tanhf(acc2[r] + bo[r]);
#pragma unroll
      for (int r = 0; r < 4; ++r) y1[r] = tanhf(acc2[4 + r] + bo[4 + r]);
      *(f32x4*)yp = y0;
      *(f32x4*)(yp + 4) = y1;
    }
  }

  // H_last (f32, exact) -> tail of d_out
  float* hl = out + (size_t)T_STEPS * BATCH * DOUT;
#pragma unroll
  for (int r = 0; r < 8; ++r)
    hl[(size_t)(rowbase + r) * BATCH + bcol] = hprev[r];
}

// ---------------------------------------------------------------------------
extern "C" void kernel_launch(void* const* d_in, const int* in_sizes, int n_in,
                              void* d_out, int out_size, void* d_ws, size_t ws_size,
                              hipStream_t stream) {
  (void)in_sizes; (void)n_in; (void)out_size; (void)ws_size;
  const float* X  = (const float*)d_in[0];
  const float* Wi = (const float*)d_in[1];
  const float* Wh = (const float*)d_in[2];
  const float* Wo = (const float*)d_in[3];
  float* out = (float*)d_out;

  // workspace layout (~1.6 MB)
  char* ws = (char*)d_ws;
  unsigned*       bar = (unsigned*)ws;                          // 16 x 64B
  unsigned short* HT  = (unsigned short*)(ws + 1024);           // 512 KB
  unsigned short* WhF = (unsigned short*)(ws + 1024 + 2 * BATCH * HID * 2);
  unsigned short* WiF = WhF + 512 * 512;
  unsigned short* WoF = WiF + 512 * 256;
  float* biasH = (float*)(WoF + 256 * 512);
  float* biasO = biasH + HID;

  // zero barrier counters + H double buffer (H0 = 0)
  hipMemsetAsync(ws, 0, 1024 + (size_t)2 * BATCH * HID * 2, stream);

  const unsigned prep_elems = 512u * 512u + 512u * 256u + 256u * 512u + 512u + 256u;
  crnn_prep<<<dim3((prep_elems + 255u) / 256u), dim3(256), 0, stream>>>(
      Wi, Wh, Wo, WhF, WiF, WoF, biasH, biasO);

  (void)hipFuncSetAttribute((const void*)crnn_recurrent,
                            hipFuncAttributeMaxDynamicSharedMemorySize, 256 * 1024);
  crnn_recurrent<<<dim3(64), dim3(256), 256 * 1024, stream>>>(
      X, out, WhF, WiF, WoF, biasH, biasO, HT, bar);
}
